// BasicBlock_8418135900537
// MI455X (gfx1250) — matrix operand order repaired and checked by reading, not compiled, not run on hardware
//
#include <hip/hip_runtime.h>

typedef __attribute__((ext_vector_type(16))) __bf16 v16bf;
typedef __attribute__((ext_vector_type(8)))  __bf16 v8bf;
typedef __attribute__((ext_vector_type(8)))  float  v8f;

#define C_DIM 128
#define NT_PER_ROW 8        // 128 / 16 output-channel tiles
#define KT_PER_ROW 4        // 128 / 32 K tiles
#define N_FRAGS (NT_PER_ROW * KT_PER_ROW)   // 32 B fragments per tile
#define W_ELEMS (C_DIM * C_DIM)             // 16384 bf16 per k-offset = 32 KB
#define WAVES_PER_BLK 8

// ---------------- init: cast x -> bf16, zero accumulator + BN stats ----------------
__global__ void k_init(const float* __restrict__ x, __bf16* __restrict__ xb,
                       float* __restrict__ acc, float* __restrict__ stats,
                       int total) {
  int i = blockIdx.x * blockDim.x + threadIdx.x;
  if (i < total) {
    xb[i]  = (__bf16)x[i];
    acc[i] = 0.0f;
  }
  if (i < 512) stats[i] = 0.0f;
}

// ---------------- repack weights into per-lane WMMA B-fragment layout ----------------
// w : [K][128(cin)][128(cout)] fp32   (einsum 'kmc,kcd->kmd')
// wp: [K][nt(8)][kt(4)][lane(32)][j(16)] bf16 with
//     kk = kt*32 + (j>>3)*16 + (lane>>4)*8 + (j&7) ,  n = nt*16 + (lane&15)
__global__ void k_repack(const float* __restrict__ w, __bf16* __restrict__ wp,
                         int total) {
  int t = blockIdx.x * blockDim.x + threadIdx.x;
  if (t >= total) return;
  int j    =  t        & 15;
  int lane = (t >> 4)  & 31;
  int kt   = (t >> 9)  & 3;
  int nt   = (t >> 11) & 7;
  int k    =  t >> 14;
  int kk = kt * 32 + ((j >> 3) << 4) + ((lane >> 4) << 3) + (j & 7);
  int n  = nt * 16 + (lane & 15);
  wp[t] = (__bf16)w[((size_t)k * C_DIM + kk) * C_DIM + n];
}

// ---------------- gather -> WMMA GEMM (B staged in LDS) -> scatter-add ----------------
// One block = one k-offset, 8 m-tiles (one per wave). Weight block lives in LDS.
// B-fragment stream is software-pipelined one step ahead of the WMMAs.
__global__ void __launch_bounds__(256)
k_conv(const __bf16* __restrict__ xb, const __bf16* __restrict__ wp,
       const int* __restrict__ idx_in, const int* __restrict__ idx_out,
       float* __restrict__ acc, int M, int blksPerK) {
  __shared__ __attribute__((aligned(32))) __bf16 sw[W_ELEMS];   // 32 KB

  const int k    = blockIdx.x / blksPerK;
  const int mt   = (blockIdx.x - k * blksPerK) * WAVES_PER_BLK + (threadIdx.x >> 5);
  const int lane = threadIdx.x & 31;
  const int mtPerK = M >> 4;

  // ---- cooperative LDS fill of this k's packed weights (2048 x uint4) ----
  {
    const uint4* src = (const uint4*)(wp + (size_t)k * W_ELEMS);
    uint4* dst = (uint4*)sw;
#pragma unroll
    for (int i = 0; i < W_ELEMS / 8 / 256; ++i)      // 8 iters @ 256 threads
      dst[i * 256 + threadIdx.x] = src[i * 256 + threadIdx.x];
  }
  __syncthreads();

  if (mt < mtPerK) {                       // wave-uniform guard: EXEC all-1 inside
    const int h    = lane >> 4;            // K-half select (A), row-half select (C/D)
    const int mloc = lane & 15;            // A row in tile / C column in tile

    // A fragments: gathered row, four 16x32 K-tiles.
    // 16-bit A 16x32 layout: lanes 0-15 hold K=[0..7]+[16..23], lanes 16-31 K=[8..15]+[24..31].
    const int arow = idx_in[(size_t)k * M + (mt << 4) + mloc];
    const __bf16* xrow = xb + (size_t)arow * C_DIM;
    union { v16bf v; v8bf h2[2]; } a[KT_PER_ROW];
#pragma unroll
    for (int kt = 0; kt < KT_PER_ROW; ++kt) {
      a[kt].h2[0] = *(const v8bf*)(xrow + kt * 32 +      h * 8);
      a[kt].h2[1] = *(const v8bf*)(xrow + kt * 32 + 16 + h * 8);
    }

    // C/D layout: VGPR i holds M=i (lanes 0-15) or M=8+i (lanes 16-31), N = lane&15.
    int ro[8];
    const int* iop = idx_out + (size_t)k * M + (mt << 4) + (h << 3);
#pragma unroll
    for (int i = 0; i < 8; ++i) ro[i] = iop[i];

    // ---- pipelined fragment stream: 32 fragments, load t+1 before WMMA t ----
    const __bf16* bbase = sw + lane * 16;
    v16bf bcur = *(const v16bf*)(bbase);
    v8f c = {};
#pragma unroll
    for (int t = 0; t < N_FRAGS; ++t) {
      v16bf bnext = bcur;
      if (t + 1 < N_FRAGS)
        bnext = *(const v16bf*)(bbase + (t + 1) * 512);   // ds_load_b128 x2, 1 step ahead
      c = __builtin_amdgcn_wmma_f32_16x16x32_bf16(false, a[t & 3].v, false, bcur,
                                                  (short)0, c, false, false);
      if ((t & 3) == 3) {                   // finished a 16x16 output tile
        int nt = t >> 2;
        float* op = acc + nt * 16 + mloc;
#pragma unroll
        for (int i = 0; i < 8; ++i) {
          __hip_atomic_fetch_add(op + (size_t)(ro[i] * C_DIM), c[i],
                                 __ATOMIC_RELAXED, __HIP_MEMORY_SCOPE_AGENT);
        }
        c = v8f{};
      }
      bcur = bnext;
    }
  }
}

// ---------------- BN statistics: per-channel sum / sum-of-squares ----------------
__global__ void k_bnstats(const float* __restrict__ acc, float* __restrict__ sums,
                          float* __restrict__ sumsq, int N) {
  int c = threadIdx.x & (C_DIM - 1);
  int rowsPerBlk = blockDim.x / C_DIM;
  int row = blockIdx.x * rowsPerBlk + (threadIdx.x >> 7);
  int stride = gridDim.x * rowsPerBlk;
  float s = 0.f, q = 0.f;
  for (; row < N; row += stride) {
    float v = acc[(size_t)row * C_DIM + c];
    s += v; q += v * v;
  }
  __hip_atomic_fetch_add(sums  + c, s, __ATOMIC_RELAXED, __HIP_MEMORY_SCOPE_AGENT);
  __hip_atomic_fetch_add(sumsq + c, q, __ATOMIC_RELAXED, __HIP_MEMORY_SCOPE_AGENT);
}

// ---------------- BN + ReLU -> bf16 activations; re-zero accumulator ----------------
__global__ void k_bn_relu_tobf16(float* __restrict__ acc, const float* __restrict__ stats,
                                 const float* __restrict__ gamma, const float* __restrict__ beta,
                                 __bf16* __restrict__ y, int N) {
  int i = blockIdx.x * blockDim.x + threadIdx.x;
  if (i >= N * C_DIM) return;
  int c = i & (C_DIM - 1);
  float invN = 1.0f / (float)N;
  float mean = stats[c] * invN;
  float var  = stats[C_DIM + c] * invN - mean * mean;
  float inv  = rsqrtf(var + 1e-5f);
  float v = (acc[i] - mean) * inv * gamma[c] + beta[c];
  v = v > 0.f ? v : 0.f;
  y[i]   = (__bf16)v;
  acc[i] = 0.0f;                 // ready for conv2 scatter-add
}

// ---------------- BN + residual + ReLU -> fp32 output ----------------
__global__ void k_bn_residual(const float* __restrict__ acc, const float* __restrict__ stats,
                              const float* __restrict__ gamma, const float* __restrict__ beta,
                              const float* __restrict__ x, float* __restrict__ out, int N) {
  int i = blockIdx.x * blockDim.x + threadIdx.x;
  if (i >= N * C_DIM) return;
  int c = i & (C_DIM - 1);
  float invN = 1.0f / (float)N;
  float mean = stats[c] * invN;
  float var  = stats[C_DIM + c] * invN - mean * mean;
  float inv  = rsqrtf(var + 1e-5f);
  float v = (acc[i] - mean) * inv * gamma[c] + beta[c] + x[i];
  out[i] = v > 0.f ? v : 0.f;
}

extern "C" void kernel_launch(void* const* d_in, const int* in_sizes, int n_in,
                              void* d_out, int out_size, void* d_ws, size_t ws_size,
                              hipStream_t stream) {
  const float* x      = (const float*)d_in[0];
  const float* w1     = (const float*)d_in[1];
  const float* gamma1 = (const float*)d_in[2];
  const float* beta1  = (const float*)d_in[3];
  const float* w2     = (const float*)d_in[4];
  const float* gamma2 = (const float*)d_in[5];
  const float* beta2  = (const float*)d_in[6];
  const int* idx_in   = (const int*)d_in[7];
  const int* idx_out  = (const int*)d_in[8];

  const int N = in_sizes[0] / C_DIM;            // 100000
  const int K = in_sizes[1] / (C_DIM * C_DIM);  // 27
  const int M = in_sizes[7] / K;                // 50000  (divisible by 16)
  const int total = N * C_DIM;

  // ---- workspace carve-up ----
  char* ws = (char*)d_ws;
  size_t nb_bf = (size_t)total * sizeof(__bf16);
  __bf16* xb   = (__bf16*)ws;  ws += nb_bf;                          // 25.6 MB
  __bf16* yb   = (__bf16*)ws;  ws += nb_bf;                          // 25.6 MB
  float*  acc  = (float*)ws;   ws += (size_t)total * sizeof(float);  // 51.2 MB
  __bf16* w1p  = (__bf16*)ws;  ws += (size_t)K * W_ELEMS * 2;        // 0.88 MB
  __bf16* w2p  = (__bf16*)ws;  ws += (size_t)K * W_ELEMS * 2;        // 0.88 MB
  float*  stats= (float*)ws;                                         // 512 floats

  const int blk = 256;
  k_init<<<(total + blk - 1) / blk, blk, 0, stream>>>(x, xb, acc, stats, total);

  const int wtotal = K * W_ELEMS;
  k_repack<<<(wtotal + blk - 1) / blk, blk, 0, stream>>>(w1, w1p, wtotal);
  k_repack<<<(wtotal + blk - 1) / blk, blk, 0, stream>>>(w2, w2p, wtotal);

  const int mtPerK   = M >> 4;                                        // 3125
  const int blksPerK = (mtPerK + WAVES_PER_BLK - 1) / WAVES_PER_BLK;  // 391
  const int convBlocks = K * blksPerK;

  // conv1 -> BN -> ReLU
  k_conv<<<convBlocks, blk, 0, stream>>>(xb, w1p, idx_in, idx_out, acc, M, blksPerK);
  k_bnstats<<<512, blk, 0, stream>>>(acc, stats + 0, stats + C_DIM, N);
  k_bn_relu_tobf16<<<(total + blk - 1) / blk, blk, 0, stream>>>(acc, stats, gamma1, beta1, yb, N);

  // conv2 -> BN -> +identity -> ReLU
  k_conv<<<convBlocks, blk, 0, stream>>>(yb, w2p, idx_in, idx_out, acc, M, blksPerK);
  k_bnstats<<<512, blk, 0, stream>>>(acc, stats + 2 * C_DIM, stats + 3 * C_DIM, N);
  k_bn_residual<<<(total + blk - 1) / blk, blk, 0, stream>>>(acc, stats + 2 * C_DIM,
                                                             gamma2, beta2, x,
                                                             (float*)d_out, N);
}
